// RotaryMultiheadAttention_20744692040030
// MI455X (gfx1250) — matrix operand order repaired and checked by reading, not compiled, not run on hardware
//
#include <hip/hip_runtime.h>
#include <hip/hip_bf16.h>
#include <math.h>

typedef __attribute__((ext_vector_type(16))) _Float16 v16h;
typedef __attribute__((ext_vector_type(8)))  _Float16 v8h;
typedef __attribute__((ext_vector_type(8)))  float    v8f;

#define BB  2
#define SS  2048
#define EE  1024
#define NHH 16
#define HD  64
#define E3  3072

__device__ __forceinline__ v16h cat8(v8h lo, v8h hi) {
  return __builtin_shufflevector(lo, hi, 0,1,2,3,4,5,6,7,8,9,10,11,12,13,14,15);
}
// A-fragment (16x32 f16, M x K): lane 0-15 row M=lane holds K kb..kb+15,
// lane 16-31 row M=lane-16 holds K kb+16..kb+31.  hi8 = (lane>=16)?8:0.
__device__ __forceinline__ v16h lda(const _Float16* row, int kb, int hi8) {
  v8h lo = *(const v8h*)(row + kb + 2 * hi8);
  v8h hi = *(const v8h*)(row + kb + 2 * hi8 + 8);
  return cat8(lo, hi);
}
// B-fragment (32x16 f16, K x N): lane 0-15 (col N=lane) holds K 0..15, lane 16-31 K 16..31.
// Caller passes p = &Brow_n[kb] + hi16 so halves are contiguous 16 elements.
__device__ __forceinline__ v16h ldb(const _Float16* p) {
  v8h lo = *(const v8h*)(p);
  v8h hi = *(const v8h*)(p + 8);
  return cat8(lo, hi);
}
#define WMMA(a,b,c) __builtin_amdgcn_wmma_f32_16x16x32_f16(false,(a),false,(b),(short)0,(c),false,false)

// ---------------- Kernel 1: fp32 -> f16 conversion of X and W ----------------
__global__ __launch_bounds__(256) void cvt_kernel(const float* __restrict__ X,
                                                  const float* __restrict__ W,
                                                  _Float16* __restrict__ Xh,
                                                  _Float16* __restrict__ Wh) {
  const int NX = BB * SS * EE;     // 4194304
  const int NW = E3 * EE;          // 3145728
  int idx = blockIdx.x * 256 + threadIdx.x;
  if (idx < NX) {
    Xh[idx] = (_Float16)X[idx];
  } else {
    int j = idx - NX;
    if (j < NW) Wh[j] = (_Float16)W[j];
  }
}

// ---------------- Kernel 2: QKV GEMM  C[4096,3072] = Xh * Wh^T + bias --------
// Block = 256 threads = 8 waves arranged 2(M) x 4(N); block tile 64x256,
// wave tile 32x64 (2 M-subtiles x 4 N-subtiles), K stepped by 32.
__global__ __launch_bounds__(256) void qkv_gemm(const _Float16* __restrict__ Xh,
                                                const _Float16* __restrict__ Wh,
                                                const float* __restrict__ bias,
                                                _Float16* __restrict__ qkvh) {
  const int wave = threadIdx.x >> 5;
  const int lane = threadIdx.x & 31;
  const int lh   = lane & 15;
  const int hi   = lane >> 4;
  const int hi8  = hi * 8;
  const int hi16 = hi * 16;

  const int bm = blockIdx.x % (4096 / 64);
  const int bn = blockIdx.x / (4096 / 64);
  const int m0 = bm * 64 + (wave >> 2) * 32;
  const int n0 = bn * 256 + (wave & 3) * 64;

  v8f zero = {};
  v8f acc[2][4];
#pragma unroll
  for (int i = 0; i < 2; ++i)
#pragma unroll
    for (int t = 0; t < 4; ++t) acc[i][t] = zero;

  const _Float16* a0r = Xh + (size_t)(m0 + lh) * EE;
  const _Float16* a1r = Xh + (size_t)(m0 + 16 + lh) * EE;

  for (int kb = 0; kb < EE; kb += 32) {
    v16h a0 = lda(a0r, kb, hi8);
    v16h a1 = lda(a1r, kb, hi8);
#pragma unroll
    for (int t = 0; t < 4; ++t) {
      const _Float16* br = Wh + (size_t)(n0 + t * 16 + lh) * EE + kb + hi16;
      v16h bf = ldb(br);
      acc[0][t] = WMMA(a0, bf, acc[0][t]);
      acc[1][t] = WMMA(a1, bf, acc[1][t]);
    }
  }
  // epilogue: C layout -> VGPR r holds row (r + 8*hi), col (n + lh)
#pragma unroll
  for (int mi = 0; mi < 2; ++mi)
#pragma unroll
    for (int t = 0; t < 4; ++t) {
      int col = n0 + t * 16 + lh;
      float bv = bias[col];
#pragma unroll
      for (int r = 0; r < 8; ++r) {
        int row = m0 + mi * 16 + r + hi8;
        qkvh[(size_t)row * E3 + col] = (_Float16)(acc[mi][t][r] + bv);
      }
    }
}

// ---------------- Kernel 3: RoPE on q,k + layout pack --------------------------
// qh,kh: [B,NH,S,64] f16 (rotary applied);  vt: [B,NH,64,S] f16 (transposed).
__global__ __launch_bounds__(256) void rope_pack(const _Float16* __restrict__ qkvh,
                                                 _Float16* __restrict__ qh,
                                                 _Float16* __restrict__ kh,
                                                 _Float16* __restrict__ vt) {
  int idx = blockIdx.x * 256 + threadIdx.x;   // B*S*NH*32 = 2097152 threads
  int dp = idx & 31;
  int h  = (idx >> 5) & 15;
  int s  = (idx >> 9) & 2047;
  int b  = idx >> 20;
  if (b >= BB) return;

  const _Float16* row = qkvh + (size_t)(b * SS + s) * E3;
  int col = h * HD + 2 * dp;
  float qe = (float)row[col],          qo = (float)row[col + 1];
  float ke = (float)row[EE + col],     ko = (float)row[EE + col + 1];
  float ve = (float)row[2 * EE + col], vo = (float)row[2 * EE + col + 1];

  float inv = __powf(10000.0f, -(float)dp * (1.0f / 32.0f));
  float ang = (float)s * inv;
  float sn, cs;
  __sincosf(ang, &sn, &cs);

  size_t o = ((size_t)(b * NHH + h) * SS + s) * HD + 2 * dp;
  qh[o]     = (_Float16)(qe * cs - qo * sn);
  qh[o + 1] = (_Float16)(qe * sn + qo * cs);
  kh[o]     = (_Float16)(ke * cs - ko * sn);
  kh[o + 1] = (_Float16)(ke * sn + ko * cs);

  size_t ov = ((size_t)(b * NHH + h) * HD + 2 * dp) * SS + s;
  vt[ov]      = (_Float16)ve;
  vt[ov + SS] = (_Float16)vo;
}

// ---------------- Kernel 4: flash attention, one 16-row q-tile per wave --------
__global__ __launch_bounds__(128) void attn_kernel(const _Float16* __restrict__ qh,
                                                   const _Float16* __restrict__ kh,
                                                   const _Float16* __restrict__ vt,
                                                   float* __restrict__ out) {
  __shared__ _Float16 Pl[4][16 * 32];   // per-wave 16x32 P tile (1 KB each)
  const int wave = threadIdx.x >> 5;
  const int lane = threadIdx.x & 31;
  const int lh   = lane & 15;
  const int hi   = lane >> 4;
  const int hi8  = hi * 8;
  const int hi16 = hi * 16;

  const int tile = blockIdx.x * 4 + wave;       // B*NH*(S/16) = 4096 tiles
  const int qt = tile & (SS / 16 - 1);
  const int bh = tile >> 7;                     // b*NH + h
  const int q0 = qt * 16;

  const _Float16* qb = qh + (size_t)bh * SS * HD;
  const _Float16* kbp = kh + (size_t)bh * SS * HD;
  const _Float16* vb = vt + (size_t)bh * HD * SS;

  // Q fragments: K-dim 64 => two k-steps, loaded once
  const _Float16* qrow = qb + (size_t)(q0 + lh) * HD;
  v16h qa0 = lda(qrow, 0, hi8);
  v16h qa1 = lda(qrow, 32, hi8);

  float m[8], l[8];
  v8f zero = {};
  v8f acc[4];
#pragma unroll
  for (int r = 0; r < 8; ++r) { m[r] = -1e30f; l[r] = 0.0f; }
#pragma unroll
  for (int t = 0; t < 4; ++t) acc[t] = zero;

  _Float16* pl = &Pl[wave][0];

  for (int kv = 0; kv < SS; kv += 32) {
    // scores S[16 x 32] = Q[16x64] * K_chunk^T  (two 16-wide N-tiles)
    v8f s0 = zero, s1 = zero;
    const _Float16* kr0 = kbp + (size_t)(kv + lh) * HD + hi16;
    const _Float16* kr1 = kbp + (size_t)(kv + 16 + lh) * HD + hi16;
    s0 = WMMA(qa0, ldb(kr0), s0);
    s0 = WMMA(qa1, ldb(kr0 + 32), s0);
    s1 = WMMA(qa0, ldb(kr1), s1);
    s1 = WMMA(qa1, ldb(kr1 + 32), s1);

    // online softmax; row M = r + 8*hi lives in VGPR r across a 16-lane group
#pragma unroll
    for (int r = 0; r < 8; ++r) {
      float a = s0[r] * 0.125f;          // 1/sqrt(64)
      float b = s1[r] * 0.125f;
      float cm = fmaxf(a, b);
      cm = fmaxf(cm, __shfl_xor(cm, 8, 16));
      cm = fmaxf(cm, __shfl_xor(cm, 4, 16));
      cm = fmaxf(cm, __shfl_xor(cm, 2, 16));
      cm = fmaxf(cm, __shfl_xor(cm, 1, 16));
      float mn = fmaxf(m[r], cm);
      float alpha = __expf(m[r] - mn);
      float pa = __expf(a - mn);
      float pb = __expf(b - mn);
      float rs = pa + pb;
      rs += __shfl_xor(rs, 8, 16);
      rs += __shfl_xor(rs, 4, 16);
      rs += __shfl_xor(rs, 2, 16);
      rs += __shfl_xor(rs, 1, 16);
      l[r] = l[r] * alpha + rs;
      m[r] = mn;
#pragma unroll
      for (int t = 0; t < 4; ++t) acc[t][r] *= alpha;
      int rr = r + hi8;
      pl[rr * 32 + lh]      = (_Float16)pa;   // C-layout -> LDS row-major
      pl[rr * 32 + 16 + lh] = (_Float16)pb;
    }
    asm volatile("" ::: "memory");   // keep LDS store->load order (same wave, HW in-order)

    // Re-read P as a 16x32 A-fragment from LDS
    const _Float16* prow = pl + lh * 32;
    v8h plo = *(const v8h*)(prow + hi16);
    v8h phi = *(const v8h*)(prow + hi16 + 8);
    v16h pf = cat8(plo, phi);
    asm volatile("" ::: "memory");

    // out[16x64] += P[16x32] * V_chunk[32x64]; B-fragments contiguous from vt
#pragma unroll
    for (int t = 0; t < 4; ++t) {
      const _Float16* vr = vb + (size_t)(t * 16 + lh) * SS + kv + hi16;
      acc[t] = WMMA(pf, ldb(vr), acc[t]);
    }
  }

  // normalize + store to out[B,S,NH,64] fp32
  const int b = bh >> 4, h = bh & 15;
#pragma unroll
  for (int r = 0; r < 8; ++r) {
    int rr = r + hi8;
    float invl = 1.0f / l[r];
    size_t base = ((size_t)(b * SS + q0 + rr) * NHH + h) * HD;
#pragma unroll
    for (int t = 0; t < 4; ++t)
      out[base + t * 16 + lh] = acc[t][r] * invl;
  }
}

extern "C" void kernel_launch(void* const* d_in, const int* in_sizes, int n_in,
                              void* d_out, int out_size, void* d_ws, size_t ws_size,
                              hipStream_t stream) {
  (void)in_sizes; (void)n_in; (void)out_size; (void)ws_size;
  const float* X    = (const float*)d_in[0];   // hidden_states [2,2048,1024]
  const float* W    = (const float*)d_in[1];   // in_proj_weight [3072,1024]
  const float* bias = (const float*)d_in[2];   // in_proj_bias [3072]
  float* out = (float*)d_out;

  char* ws = (char*)d_ws;
  _Float16* Xh   = (_Float16*)(ws);                 //  8,388,608 B
  _Float16* Wh   = (_Float16*)(ws + 8388608);       //  6,291,456 B
  _Float16* qkvh = (_Float16*)(ws + 14680064);      // 25,165,824 B
  _Float16* qhp  = (_Float16*)(ws + 39845888);      //  8,388,608 B
  _Float16* khp  = (_Float16*)(ws + 48234496);      //  8,388,608 B
  _Float16* vtp  = (_Float16*)(ws + 56623104);      //  8,388,608 B (ends 65,011,712)

  cvt_kernel<<<28672, 256, 0, stream>>>(X, W, Xh, Wh);
  qkv_gemm<<<768, 256, 0, stream>>>(Xh, Wh, bias, qkvh);
  rope_pack<<<8192, 256, 0, stream>>>(qkvh, qhp, khp, vtp);
  attn_kernel<<<1024, 128, 0, stream>>>(qhp, khp, vtp, out);
}
